// SelfAttention_22127671509005
// MI455X (gfx1250) — matrix operand order repaired and checked by reading, not compiled, not run on hardware
//
#include <hip/hip_runtime.h>
#include <hip/hip_bf16.h>

typedef unsigned short u16;
typedef __attribute__((ext_vector_type(16))) __bf16 v16bf;
typedef __attribute__((ext_vector_type(8)))  float  v8f;

#define BATCH 8
#define CIN   256
#define HW    4096
#define DQK   128
#define DVV   256
#define DVX   272   // 256 value rows + 16 identical w-rows for the lsum WMMA
#define PADH  66
#define PADW  72    // 144B rows -> 16B aligned for async b128
#define CTR   2304  // C*9 contraction, reordered as tap*256 + c
// exp2-domain scales
#define SQK_SCALE  (2.0f * 1.44269504088896340736f / 11.3137084989847603904f)
#define BIAS_SCALE (1.44269504088896340736f / 11.3137084989847603904f)

union Frag16 { v16bf v; u16 s[16]; uint4 q[2]; };
union Pack8  { u16 s[8]; uint4 q; };

static __device__ inline u16 f2bf(float f) {              // RNE
    unsigned u = __float_as_uint(f);
    unsigned r = u + 0x7FFFu + ((u >> 16) & 1u);
    return (u16)(r >> 16);
}
static __device__ inline float bf2f(u16 s) { return __uint_as_float(((unsigned)s) << 16); }
static __device__ inline unsigned packbf2(float a, float b) {   // round-half-up pair
    unsigned ua = __float_as_uint(a) + 0x8000u;
    unsigned ub = __float_as_uint(b) + 0x8000u;
    return (ua >> 16) | (ub & 0xFFFF0000u);
}
static __device__ inline v8f wmma_bf16(const Frag16& a, const Frag16& b, v8f c) {
    return __builtin_amdgcn_wmma_f32_16x16x32_bf16(false, a.v, false, b.v, (short)0, c, false, false);
}
static __device__ inline unsigned lds_off(const void* p) {
    return (unsigned)(unsigned long long)p;      // LDS offset = low 32 bits of the address
}
static __device__ inline void async_b128(unsigned lds, unsigned voff, unsigned long long base) {
    asm volatile("global_load_async_to_lds_b128 %0, %1, %2"
                 :: "v"(lds), "v"(voff), "s"(base) : "memory");
}

// ---------------------------------------------------------------------------
// Prep: padded + bf16-split x:  xpad[b][c][66][72], xph + xpl, zero borders.
// ---------------------------------------------------------------------------
__global__ __launch_bounds__(256) void padsplit_x_kernel(
    const float* __restrict__ x, u16* __restrict__ xph, u16* __restrict__ xpl)
{
    int idx = blockIdx.x * 256 + threadIdx.x;        // b*256*66 + c*66 + hh
    int hh = idx % PADH, bc = idx / PADH;            // bc = b*256 + c
    const float* src = x + ((size_t)bc * 64 + (hh - 1)) * 64;
    u16* dh = xph + ((size_t)bc * PADH + hh) * PADW;
    u16* dl = xpl + ((size_t)bc * PADH + hh) * PADW;
    bool hv = (hh >= 1) && (hh <= 64);
    for (int ww = 0; ww < PADW; ++ww) {
        float v = (hv && ww >= 1 && ww <= 64) ? src[ww - 1] : 0.0f;
        u16 h = f2bf(v);
        dh[ww] = h;
        dl[ww] = f2bf(v - bf2f(h));
    }
}

// Prep: split + reorder weights: dst[oc][t*256+c] = w[oc][c][t] (OIHW, t = ky*3+kx)
__global__ __launch_bounds__(256) void split_w_kernel(
    const float* __restrict__ wgt, u16* __restrict__ wh, u16* __restrict__ wl, int total)
{
    int idx = blockIdx.x * 256 + threadIdx.x;
    if (idx >= total) return;
    int oc = idx / CTR, i = idx % CTR;
    int t = i >> 8, c = i & 255;
    float v = wgt[((size_t)oc * 256 + c) * 9 + t];
    u16 h = f2bf(v);
    wh[idx] = h;
    wl[idx] = f2bf(v - bf2f(h));
}

// ---------------------------------------------------------------------------
// Conv 3x3 as implicit GEMM on WMMA, bf16-split (wh*xh + wh*xl + wl*xh), fp32 acc.
// Block: 32 oc x 128 px (8 waves: 2 oc-rows x 4 px-cols, 2 accs each).
// 72 chunks of 32 contraction (tap-outer => fixed kx,ky per chunk), async-staged.
// mode 0: key conv -> bf16 kq[b][j][128]
// mode 1: val conv -> fp32 out[b][oc][j] + bf16 vt[b][oc][j]*w_j
// ---------------------------------------------------------------------------
__global__ __launch_bounds__(256) void conv_wmma_kernel(
    const u16* __restrict__ xph, const u16* __restrict__ xpl,
    const u16* __restrict__ wh,  const u16* __restrict__ wl,
    const float* __restrict__ bias, int mode,
    u16* __restrict__ bf_out, float* __restrict__ f_out,
    const float* __restrict__ bias2)
{
    const int b   = blockIdx.z;
    const int oc0 = blockIdx.y * 32;
    const int px0 = blockIdx.x * 128;
    const int h0  = px0 >> 6;                        // two output image rows
    const int tid = threadIdx.x;
    const int wave = tid >> 5, lane = tid & 31;
    const int lhalf = lane >> 4, l16 = lane & 15;
    const int wrow16 = (wave >> 2) * 16;             // wave's oc sub-block
    const int pxw = px0 + (wave & 3) * 32;           // wave's 32 px

    __shared__ u16 Xs[2][2][32 * 2 * PADW];          // [buf][plane][c][row][col]
    __shared__ u16 Ws[2][2][32 * 32];                // [buf][plane][oc][ci]

    const unsigned long long xb[2] = {
        (unsigned long long)(xph + (size_t)b * 256 * PADH * PADW),
        (unsigned long long)(xpl + (size_t)b * 256 * PADH * PADW) };
    const unsigned ldsX0 = lds_off(&Xs[0][0][0]);
    const unsigned ldsW0 = lds_off(&Ws[0][0][0]);

    // ---- hoisted stage-slot decomposition (loop-invariant per thread) ----
    const int nslot = (tid < 64) ? 3 : 2;            // i = tid + 256*s < 576
    unsigned xlds[3], xvoffc[3];
#pragma unroll
    for (int s = 0; s < 3; ++s) {
        int i = tid + 256 * s;
        int row = i / 9, part = i % 9;               // row = cl*2+ry
        int cl = row >> 1, ry = row & 1;
        xlds[s]   = (unsigned)(row * PADW + part * 8) * 2u;
        xvoffc[s] = ((unsigned)(cl * PADH + ry) * PADW + part * 8u) * 2u;
    }
    const unsigned long long wbase = (tid >> 7) ? (unsigned long long)wl
                                                : (unsigned long long)wh;
    const unsigned wpl   = (unsigned)(tid >> 7);
    const unsigned wlds  = ((unsigned)(((tid & 127) >> 2) * 32 + (tid & 3) * 8)) * 2u;
    const unsigned wvoffc = ((unsigned)(oc0 + ((tid & 127) >> 2)) * CTR
                           + (unsigned)(tid & 3) * 8u) * 2u;

    auto stage = [&](int buf, int it) {
        int t = it >> 3, c0 = (it & 7) * 32;
        int ky = t / 3;
        unsigned itoff = ((unsigned)(c0 * PADH + h0 + ky) * PADW) * 2u;
#pragma unroll
        for (int pl = 0; pl < 2; ++pl) {
            unsigned lx = ldsX0 + buf * (2 * 32 * 2 * PADW * 2) + pl * (32 * 2 * PADW * 2);
#pragma unroll
            for (int s = 0; s < 3; ++s)
                if (s < nslot)
                    async_b128(lx + xlds[s], itoff + xvoffc[s], xb[pl]);
        }
        unsigned lw = ldsW0 + buf * (2 * 1024 * 2) + wpl * (1024 * 2);
        async_b128(lw + wlds, wvoffc + (unsigned)(t * 256 + c0) * 2u, wbase);
    };

    // ---- hoisted B-gather column bases ----
    int colbase[2];
#pragma unroll
    for (int sub = 0; sub < 2; ++sub) {
        int pxl = (pxw - px0) + sub * 16 + l16;      // 0..127
        colbase[sub] = (pxl >> 6) * PADW + (pxl & 63);
    }
    const int grow = 16 * lhalf * 2 * PADW;          // gather channel-half base

    v8f acc[2] = { (v8f)(0.0f), (v8f)(0.0f) };

    stage(0, 0);
    asm volatile("s_wait_asynccnt 0x0" ::: "memory");
    __syncthreads();

    for (int it = 0; it < 72; ++it) {
        const int cur = it & 1;
        if (it + 1 < 72) stage(cur ^ 1, it + 1);
        const int kx = (it >> 3) % 3;

        // A fragments: wave's 16 oc rows, 32 ci
        const u16* wsh = &Ws[cur][0][0];
        const u16* wsl = &Ws[cur][1][0];
        Frag16 ah, al;
        {
            int rowb = (wrow16 + l16) * 32;
            ah.q[0] = *(const uint4*)(wsh + rowb + lhalf * 16);
            ah.q[1] = *(const uint4*)(wsh + rowb + lhalf * 16 + 8);
            al.q[0] = *(const uint4*)(wsl + rowb + lhalf * 16);
            al.q[1] = *(const uint4*)(wsl + rowb + lhalf * 16 + 8);
        }
        // B fragments: gather (lane N=px, K elem e -> channel 16*lhalf+e), kx folded in
        const u16* xsh = &Xs[cur][0][0];
        const u16* xsl = &Xs[cur][1][0];
#pragma unroll
        for (int sub = 0; sub < 2; ++sub) {
            int base = grow + colbase[sub] + kx;
            Frag16 bh, bl;
#pragma unroll
            for (int e = 0; e < 16; ++e) {
                int a = base + e * 2 * PADW;
                bh.s[e] = xsh[a];
                bl.s[e] = xsl[a];
            }
            acc[sub] = wmma_bf16(ah, bh, acc[sub]);
            acc[sub] = wmma_bf16(ah, bl, acc[sub]);
            acc[sub] = wmma_bf16(al, bh, acc[sub]);
        }

        asm volatile("s_wait_asynccnt 0x0" ::: "memory");
        __syncthreads();
    }

    float bv[8];
#pragma unroll
    for (int r = 0; r < 8; ++r) bv[r] = bias[oc0 + wrow16 + r + 8 * lhalf];

#pragma unroll
    for (int sub = 0; sub < 2; ++sub) {
        int j = pxw + sub * 16 + l16;
        if (mode == 0) {
            Pack8 pk;
#pragma unroll
            for (int r = 0; r < 8; ++r) pk.s[r] = f2bf(acc[sub][r] + bv[r]);
            ((uint4*)bf_out)[((size_t)(b * HW + j) * DQK + oc0 + wrow16 + 8 * lhalf) >> 3] = pk.q;
        } else {
            float wj = __builtin_amdgcn_exp2f(-bias2[b * HW + j]);
#pragma unroll
            for (int r = 0; r < 8; ++r) {
                int oc = oc0 + wrow16 + r + 8 * lhalf;
                float v = acc[sub][r] + bv[r];
                f_out[(size_t)(b * 512 + oc) * HW + j] = v;
                bf_out[(size_t)(b * DVX + oc) * HW + j] = f2bf(v * wj);
            }
        }
    }
}

// bias2[j] = |k_j|^2 * log2e/sqrt(128) from bf16 k; writes 16 identical w-rows of vt.
__global__ __launch_bounds__(256) void sumsq_kernel(const u16* __restrict__ kq,
                                                    float* __restrict__ bias2,
                                                    u16* __restrict__ vt)
{
    int idx = blockIdx.x * 256 + threadIdx.x;
    const u16* row = kq + (size_t)idx * DQK;
    float s = 0.0f;
#pragma unroll 8
    for (int i = 0; i < DQK; ++i) { float f = bf2f(row[i]); s = fmaf(f, f, s); }
    float b2 = s * BIAS_SCALE;
    bias2[idx] = b2;
    u16 wb = f2bf(__builtin_amdgcn_exp2f(-b2));
    int b = idx >> 12, j = idx & (HW - 1);
#pragma unroll
    for (int r = 0; r < 16; ++r)
        vt[(size_t)(b * DVX + 256 + r) * HW + j] = wb;
}

// ---------------------------------------------------------------------------
// Flash attention: fixed-shift softmax, w-folded bias, lsum via 17th WMMA,
// async double-buffered tiles.
// ---------------------------------------------------------------------------
__global__ __launch_bounds__(256) void attn_kernel(
    const u16* __restrict__ kq, const u16* __restrict__ vt,
    const float* __restrict__ bias2, float* __restrict__ out)
{
    const int b    = blockIdx.y;
    const int q0b  = blockIdx.x * 128;
    const int tid  = threadIdx.x;
    const int wave = tid >> 5, lane = tid & 31;
    const int lhalf = lane >> 4, l16 = lane & 15;
    const int q0w = q0b + wave * 16;

    __shared__ u16 Kt[2][32 * DQK];
    __shared__ u16 Vs[2][DVX * 32];
    __shared__ u16 Pt[8][32 * 16];

    Frag16 aQ[4];
    const u16* qrow = kq + (size_t)(b * HW + q0w + l16) * DQK;
#pragma unroll
    for (int f = 0; f < 4; ++f) {
        aQ[f].q[0] = *(const uint4*)(qrow + f * 32 + lhalf * 16);
        aQ[f].q[1] = *(const uint4*)(qrow + f * 32 + lhalf * 16 + 8);
    }

    float mq[8];
#pragma unroll
    for (int r = 0; r < 8; ++r) mq[r] = bias2[b * HW + q0w + r + 8 * lhalf];

    v8f O[17];
#pragma unroll
    for (int i = 0; i < 17; ++i) O[i] = (v8f)(0.0f);

    const unsigned long long kbase = (unsigned long long)(kq + (size_t)b * HW * DQK);
    const unsigned long long vbase = (unsigned long long)(vt + (size_t)b * DVX * HW);
    const unsigned ldsK0 = lds_off(&Kt[0][0]);
    const unsigned ldsV0 = lds_off(&Vs[0][0]);
    const int NT = HW / 32;

    auto stage = [&](int buf, int t) {
        unsigned lk = ldsK0 + buf * (32 * DQK * 2);
#pragma unroll
        for (int i = tid; i < 512; i += 256)
            async_b128(lk + i * 16u, (unsigned)(t * 512 + i) * 16u, kbase);
        unsigned lv = ldsV0 + buf * (DVX * 32 * 2);
        for (int i = tid; i < DVX * 4; i += 256) {
            int row = i >> 2, part = i & 3;
            async_b128(lv + i * 16u, (unsigned)(row * (HW / 8) + t * 4 + part) * 16u, vbase);
        }
        if (t + 2 < NT) {
            __builtin_prefetch((const char*)kbase + (size_t)(t + 2) * 8192, 0, 0);
            __builtin_prefetch((const char*)vbase + (size_t)tid * 8192 + (size_t)(t + 2) * 64, 0, 0);
        }
    };

    stage(0, 0);
    asm volatile("s_wait_asynccnt 0x0" ::: "memory");
    __syncthreads();

    for (int t = 0; t < NT; ++t) {
        const int cur = t & 1;
        if (t + 1 < NT) stage(cur ^ 1, t + 1);

        const uint4* kb = (const uint4*)Kt[cur];
        Frag16 bK[8];
#pragma unroll
        for (int f = 0; f < 4; ++f) {
            int off0 = (l16 * DQK + f * 32 + lhalf * 16) >> 3;
            int off1 = ((16 + l16) * DQK + f * 32 + lhalf * 16) >> 3;
            bK[2 * f + 0].q[0] = kb[off0]; bK[2 * f + 0].q[1] = kb[off0 + 1];
            bK[2 * f + 1].q[0] = kb[off1]; bK[2 * f + 1].q[1] = kb[off1 + 1];
        }
        v8f S0 = (v8f)(0.0f), S1 = (v8f)(0.0f);
#pragma unroll
        for (int f = 0; f < 4; ++f) {
            S0 = wmma_bf16(aQ[f], bK[2 * f + 0], S0);
            S1 = wmma_bf16(aQ[f], bK[2 * f + 1], S1);
        }

        float pa[8], pb[8];
#pragma unroll
        for (int r = 0; r < 8; ++r) {
            pa[r] = __builtin_amdgcn_exp2f(fmaf(S0[r], SQK_SCALE, -mq[r]));
            pb[r] = __builtin_amdgcn_exp2f(fmaf(S1[r], SQK_SCALE, -mq[r]));
        }

        u16* pw = Pt[wave];
        uint4 w0, w1;
        w0.x = packbf2(pa[0], pa[1]); w0.y = packbf2(pa[2], pa[3]);
        w0.z = packbf2(pa[4], pa[5]); w0.w = packbf2(pa[6], pa[7]);
        w1.x = packbf2(pb[0], pb[1]); w1.y = packbf2(pb[2], pb[3]);
        w1.z = packbf2(pb[4], pb[5]); w1.w = packbf2(pb[6], pb[7]);
        *(uint4*)(pw + l16 * 16 + 8 * lhalf)        = w0;
        *(uint4*)(pw + (16 + l16) * 16 + 8 * lhalf) = w1;
        asm volatile("s_wait_dscnt 0x0" ::: "memory");

        Frag16 bP;
#pragma unroll
        for (int e = 0; e < 16; ++e)
            bP.s[e] = pw[(8 * lhalf + e + (e & 8)) * 16 + l16];

        const uint4* vb = (const uint4*)Vs[cur];
        Frag16 aV[2];
        {
            int off = (l16 * 32 + lhalf * 8) >> 3;
            aV[0].q[0] = vb[off]; aV[0].q[1] = vb[off + 2];
        }
#pragma unroll
        for (int cB = 0; cB < 17; ++cB) {
            if (cB < 16) {
                int off = (((cB + 1) * 16 + l16) * 32 + lhalf * 8) >> 3;
                aV[(cB + 1) & 1].q[0] = vb[off];
                aV[(cB + 1) & 1].q[1] = vb[off + 2];
            }
            O[cB] = wmma_bf16(aV[cB & 1], bP, O[cB]);
        }

        asm volatile("s_wait_asynccnt 0x0" ::: "memory");
        __syncthreads();
    }

    const float invq = 1.0f / O[16][0];
#pragma unroll
    for (int cB = 0; cB < 16; ++cB)
#pragma unroll
        for (int r = 0; r < 8; ++r) {
            int vd = cB * 16 + r + 8 * lhalf;
            out[(size_t)(b * 512 + 256 + vd) * HW + q0w + l16] = O[cB][r] * invq;
        }
}

// ---------------------------------------------------------------------------
extern "C" void kernel_launch(void* const* d_in, const int* in_sizes, int n_in,
                              void* d_out, int out_size, void* d_ws, size_t ws_size,
                              hipStream_t stream)
{
    const float* x     = (const float*)d_in[0];
    const float* key_w = (const float*)d_in[1];
    const float* key_b = (const float*)d_in[2];
    const float* val_w = (const float*)d_in[3];
    const float* val_b = (const float*)d_in[4];
    float* out = (float*)d_out;

    // workspace (~69 MB): kq | vt | bias2 | xpad_h | xpad_l | wk_h/l | wv_h/l
    char* p = (char*)d_ws;
    u16*   kq    = (u16*)p;            p += (size_t)BATCH * HW * DQK * 2;
    u16*   vt    = (u16*)p;            p += (size_t)BATCH * DVX * HW * 2;
    float* bias2 = (float*)p;          p += (size_t)BATCH * HW * 4;
    u16*   xph   = (u16*)p;            p += (size_t)BATCH * CIN * PADH * PADW * 2;
    u16*   xpl   = (u16*)p;            p += (size_t)BATCH * CIN * PADH * PADW * 2;
    u16*   wkh   = (u16*)p;            p += (size_t)DQK * CTR * 2;
    u16*   wkl   = (u16*)p;            p += (size_t)DQK * CTR * 2;
    u16*   wvh   = (u16*)p;            p += (size_t)DVV * CTR * 2;
    u16*   wvl   = (u16*)p;            p += (size_t)DVV * CTR * 2;

    padsplit_x_kernel<<<(BATCH * CIN * PADH) / 256, 256, 0, stream>>>(x, xph, xpl);
    split_w_kernel<<<(DQK * CTR + 255) / 256, 256, 0, stream>>>(key_w, wkh, wkl, DQK * CTR);
    split_w_kernel<<<(DVV * CTR + 255) / 256, 256, 0, stream>>>(val_w, wvh, wvl, DVV * CTR);

    conv_wmma_kernel<<<dim3(HW / 128, DQK / 32, BATCH), 256, 0, stream>>>(
        xph, xpl, wkh, wkl, key_b, 0, kq, nullptr, nullptr);
    sumsq_kernel<<<(BATCH * HW) / 256, 256, 0, stream>>>(kq, bias2, vt);
    conv_wmma_kernel<<<dim3(HW / 128, DVV / 32, BATCH), 256, 0, stream>>>(
        xph, xpl, wvh, wvl, val_b, 1, vt, out, bias2);
    attn_kernel<<<dim3(HW / 128, BATCH), 256, 0, stream>>>(kq, vt, bias2, out);
}
